// MultiHeadAttention_39711267619157
// MI455X (gfx1250) — compile-verified, matrix operand order checked by reading, not run on hardware
//
#include <hip/hip_runtime.h>

// ---------------------------------------------------------------------------
// MI455X (gfx1250) fused MHA: QKV GEMM -> RMSNorm+RoPE -> flash attention
// (K@Q^T trick for shuffle-free P transpose) -> gate -> out GEMM.
// All matmuls via v_wmma_f32_16x16x32_bf16 (wave32).
// ---------------------------------------------------------------------------

#define B_      2
#define S_      2048
#define DMODEL  1024
#define NH      16
#define NKV     4
#define DK      64
#define QKV_N   1536
#define TOKENS  (B_ * S_)        // 4096

typedef __bf16 bf16x16 __attribute__((ext_vector_type(16)));
typedef float  v8f     __attribute__((ext_vector_type(8)));
typedef unsigned short u16;
typedef unsigned int   u32;

union FragAB {
    uint4   q[2];
    u32     u[8];
    bf16x16 h;
};

__device__ __forceinline__ u16 f2bf(float f) {
    u32 u = __float_as_uint(f);
    u += 0x7FFFu + ((u >> 16) & 1u);   // round-to-nearest-even
    return (u16)(u >> 16);
}
__device__ __forceinline__ u32 pack_bf(float lo, float hi) {
    return (u32)f2bf(lo) | ((u32)f2bf(hi) << 16);
}

// Load a 16x32 bf16 operand fragment from a row-major [rows, ld] bf16 matrix.
// CDNA5 16-bit A/B layout: lane = row0 + lane%16; K-half = (lane/16)*8;
// V0..V3 cover K = khalf+0..7, V4..V7 cover K = 16+khalf+0..7 (pairs packed).
__device__ __forceinline__ bf16x16 load_frag(const u16* __restrict__ base,
                                             int ld, int row0, int k0) {
    const int lane = threadIdx.x & 31;
    const u16* p = base + (size_t)(row0 + (lane & 15)) * ld + k0 + ((lane >> 4) << 3);
    FragAB f;
    f.q[0] = *(const uint4*)(p);
    f.q[1] = *(const uint4*)(p + 16);
    return f.h;
}

#define WMMA_BF16(A, Bv, Cv) \
    __builtin_amdgcn_wmma_f32_16x16x32_bf16(false, (A), false, (Bv), (short)0, (Cv), false, false)

// ---------------------------------------------------------------------------
// f32 -> bf16 conversion (4 elems / thread)
// ---------------------------------------------------------------------------
__global__ void cvt_f32_bf16(const float* __restrict__ src, u16* __restrict__ dst, int n4) {
    int i = blockIdx.x * blockDim.x + threadIdx.x;
    if (i >= n4) return;
    float4 v = ((const float4*)src)[i];
    ((uint2*)dst)[i] = make_uint2(pack_bf(v.x, v.y), pack_bf(v.z, v.w));
}

// ---------------------------------------------------------------------------
// Gate: sigmoid(x[:, :12] @ gate_w[h, :])  -> gate[token*NH + h]
// ---------------------------------------------------------------------------
__global__ void gate_kernel(const float* __restrict__ x, const float* __restrict__ gw,
                            float* __restrict__ gate) {
    int tid = blockIdx.x * blockDim.x + threadIdx.x;
    if (tid >= TOKENS * NH) return;
    int t = tid >> 4, h = tid & 15;
    const float* xr = x + (size_t)t * DMODEL;
    const float* wr = gw + h * 12;
    float d = 0.f;
#pragma unroll
    for (int g = 0; g < 12; ++g) d += xr[g] * wr[g];
    gate[tid] = 1.f / (1.f + __expf(-d));
}

// ---------------------------------------------------------------------------
// GEMM: C[M,N] f32 = A[M,K] bf16 @ B[N,K]^T bf16   (row-major A and Bw)
// Wave tile 32x64 (2x4 WMMA frags), block = 4 waves -> 64x128.
// ---------------------------------------------------------------------------
__global__ void __launch_bounds__(128) gemm_bf16(const u16* __restrict__ A,
                                                 const u16* __restrict__ Bw,
                                                 float* __restrict__ C,
                                                 int M, int N, int K) {
    const int lane = threadIdx.x & 31;
    const int wave = threadIdx.x >> 5;
    const int m0 = blockIdx.y * 64 + (wave & 1) * 32;
    const int n0 = blockIdx.x * 128 + (wave >> 1) * 64;

    v8f c[2][4] = {};
    for (int k0 = 0; k0 < K; k0 += 32) {
        bf16x16 a0 = load_frag(A, K, m0,      k0);
        bf16x16 a1 = load_frag(A, K, m0 + 16, k0);
        bf16x16 b0 = load_frag(Bw, K, n0,      k0);
        bf16x16 b1 = load_frag(Bw, K, n0 + 16, k0);
        bf16x16 b2 = load_frag(Bw, K, n0 + 32, k0);
        bf16x16 b3 = load_frag(Bw, K, n0 + 48, k0);
        c[0][0] = WMMA_BF16(a0, b0, c[0][0]);
        c[0][1] = WMMA_BF16(a0, b1, c[0][1]);
        c[0][2] = WMMA_BF16(a0, b2, c[0][2]);
        c[0][3] = WMMA_BF16(a0, b3, c[0][3]);
        c[1][0] = WMMA_BF16(a1, b0, c[1][0]);
        c[1][1] = WMMA_BF16(a1, b1, c[1][1]);
        c[1][2] = WMMA_BF16(a1, b2, c[1][2]);
        c[1][3] = WMMA_BF16(a1, b3, c[1][3]);
    }
    // C frag: lane holds col N = lane%16; VGPR v holds row M = v + (lane/16)*8
    const int col = lane & 15;
    const int hs  = (lane >> 4) << 3;
#pragma unroll
    for (int i = 0; i < 2; ++i)
#pragma unroll
        for (int j = 0; j < 4; ++j)
#pragma unroll
            for (int v = 0; v < 8; ++v)
                C[(size_t)(m0 + i * 16 + hs + v) * N + (n0 + j * 16 + col)] = c[i][j][v];
}

// ---------------------------------------------------------------------------
// RMSNorm + RoPE for Q/K heads; V copy-transpose to [d, s].
// One wave per (token, head-instance); lane i owns interleaved pair (2i, 2i+1).
// Task layout per token: r=0..15 Q heads, 16..19 K heads, 20..23 V heads.
// ---------------------------------------------------------------------------
__global__ void __launch_bounds__(256) norm_rope_kernel(const float* __restrict__ qkv,
                                                        const float* __restrict__ qw,
                                                        const float* __restrict__ kw,
                                                        u16* __restrict__ Qb,
                                                        u16* __restrict__ Kb,
                                                        u16* __restrict__ Vt) {
    const int lane = threadIdx.x & 31;
    const int wave = threadIdx.x >> 5;
    const int task = blockIdx.x * 8 + wave;
    const int token = task / 24;
    const int r = task - token * 24;
    const int b = token >> 11, s = token & 2047;
    const int i2 = lane * 2;

    if (r < 20) {
        const int h = (r < 16) ? r : (r - 16);
        const float* base = qkv + (size_t)token * QKV_N + ((r < 16) ? h * 64 : 1024 + h * 64);
        float x0 = base[i2], x1 = base[i2 + 1];
        float ss = x0 * x0 + x1 * x1;
#pragma unroll
        for (int msk = 16; msk >= 1; msk >>= 1) ss += __shfl_xor(ss, msk, 32);
        float rinv = rsqrtf(ss * (1.f / 64.f) + 1.1920929e-7f);
        const float* nw = (r < 16) ? qw : kw;
        float y0 = x0 * rinv * nw[i2];
        float y1 = x1 * rinv * nw[i2 + 1];
        // RoPE: inv_freq = 10000^(-2i/64) = exp(-i * ln(10000)/32)
        float inv = __expf(-(float)lane * (9.210340371976184f / 32.f));
        float ang = (float)s * inv;
        float cs = __cosf(ang), sn = __sinf(ang);
        float o0 = y0 * cs - y1 * sn;
        float o1 = y1 * cs + y0 * sn;
        u16* dst = (r < 16) ? (Qb + ((size_t)(b * NH + h) * S_ + s) * DK)
                            : (Kb + ((size_t)(b * NKV + h) * S_ + s) * DK);
        *(u32*)(dst + i2) = pack_bf(o0, o1);
    } else {
        const int h = r - 20;
        const float* base = qkv + (size_t)token * QKV_N + 1280 + h * 64;
        float v0 = base[i2], v1 = base[i2 + 1];
        u16* dst = Vt + ((size_t)(b * NKV + h) * DK) * S_ + s;
        dst[(size_t)i2 * S_]       = f2bf(v0);
        dst[(size_t)(i2 + 1) * S_] = f2bf(v1);
    }
}

// ---------------------------------------------------------------------------
// Flash attention with transposed scores: St = K @ Q^T so the St C-fragment
// (lane = query col, kv rows along VGPRs) converts to the P A-fragment for
// P@V by a pure per-lane bf16 pack. Online softmax, causal, 32-key blocks.
// Block = 4 waves, each wave owns a 16-query tile; grid (S/64, B*NH).
// ---------------------------------------------------------------------------
__global__ void __launch_bounds__(128) attn_kernel(const u16* __restrict__ Qb,
                                                   const u16* __restrict__ Kb,
                                                   const u16* __restrict__ Vt,
                                                   const float* __restrict__ gate,
                                                   u16* __restrict__ Ob) {
    const int lane = threadIdx.x & 31;
    const int wave = threadIdx.x >> 5;
    const int bh = blockIdx.y;
    const int b = bh >> 4, h = bh & 15;
    const int kvh = h >> 2;                       // repeat_interleave(GROUPS=4)
    const int qw = blockIdx.x * 64 + wave * 16;   // this wave's query base

    const u16* Qp = Qb + (size_t)(b * NH + h) * S_ * DK;
    const u16* Kp = Kb + (size_t)(b * NKV + kvh) * S_ * DK;
    const u16* Vp = Vt + (size_t)(b * NKV + kvh) * DK * S_;

    // Q B-fragments (lane = query col), loop-invariant
    bf16x16 q_lo = load_frag(Qp, DK, qw, 0);
    bf16x16 q_hi = load_frag(Qp, DK, qw, 32);

    v8f o[4] = {};                 // O^ frags: lane=d col, VGPR=v -> query v+half8
    float m = -1e30f, l = 0.f;     // stats for query = lane%16 (dup across halves)
    const int qg = qw + (lane & 15);
    const int half8 = (lane >> 4) << 3;

    for (int kb = 0; kb < qw + 16; kb += 32) {
        // K A-fragments: rows = kv
        bf16x16 k0a = load_frag(Kp, DK, kb,      0);
        bf16x16 k0b = load_frag(Kp, DK, kb,      32);
        bf16x16 k1a = load_frag(Kp, DK, kb + 16, 0);
        bf16x16 k1b = load_frag(Kp, DK, kb + 16, 32);

        v8f s_lo = {}, s_hi = {};
        s_lo = WMMA_BF16(k0a, q_lo, s_lo);
        s_lo = WMMA_BF16(k0b, q_hi, s_lo);
        s_hi = WMMA_BF16(k1a, q_lo, s_hi);
        s_hi = WMMA_BF16(k1b, q_hi, s_hi);

        // scale + causal mask; this lane sees kv = kb+half8+v and kb+16+half8+v
        float p[16];
        float mb = -1e30f;
#pragma unroll
        for (int v = 0; v < 8; ++v) {
            float a = s_lo[v] * 0.125f;
            if (kb + half8 + v > qg) a = -1e30f;
            p[v] = a; mb = fmaxf(mb, a);
            float c2 = s_hi[v] * 0.125f;
            if (kb + 16 + half8 + v > qg) c2 = -1e30f;
            p[8 + v] = c2; mb = fmaxf(mb, c2);
        }
        mb = fmaxf(mb, __shfl_xor(mb, 16, 32));   // merge kv halves
        float mnew = fmaxf(m, mb);
        float corr = __expf(m - mnew);
        float rs = 0.f;
#pragma unroll
        for (int v = 0; v < 16; ++v) { p[v] = __expf(p[v] - mnew); rs += p[v]; }
        rs += __shfl_xor(rs, 16, 32);
        l = l * corr + rs;
        m = mnew;

        // P^T C-frag -> P A-frag: pure per-lane pack (kv pairs along VGPRs)
        FragAB pf;
#pragma unroll
        for (int v = 0; v < 4; ++v) {
            pf.u[v]     = pack_bf(p[2 * v],     p[2 * v + 1]);
            pf.u[4 + v] = pack_bf(p[8 + 2 * v], p[8 + 2 * v + 1]);
        }

        // rescale O: factor per query row v+half8 (bpermute broadcast)
        float cv[8];
#pragma unroll
        for (int v = 0; v < 8; ++v) cv[v] = __shfl(corr, half8 + v, 32);
#pragma unroll
        for (int t = 0; t < 4; ++t)
#pragma unroll
            for (int v = 0; v < 8; ++v) o[t][v] *= cv[v];

        // V B-fragments from Vt[d, s]: lane = d col, kv along VGPRs (contiguous)
        bf16x16 v0 = load_frag(Vp, S_, 0,  kb);
        bf16x16 v1 = load_frag(Vp, S_, 16, kb);
        bf16x16 v2 = load_frag(Vp, S_, 32, kb);
        bf16x16 v3 = load_frag(Vp, S_, 48, kb);
        o[0] = WMMA_BF16(pf.h, v0, o[0]);
        o[1] = WMMA_BF16(pf.h, v1, o[1]);
        o[2] = WMMA_BF16(pf.h, v2, o[2]);
        o[3] = WMMA_BF16(pf.h, v3, o[3]);
    }

    // epilogue: 1/l and gate per query row, write bf16 [token, h*64+d]
    float linv = 1.f / l;
    float fv[8];
#pragma unroll
    for (int v = 0; v < 8; ++v) fv[v] = __shfl(linv, half8 + v, 32);
    const int col = lane & 15;
#pragma unroll
    for (int v = 0; v < 8; ++v) {
        int sq = qw + half8 + v;
        float g = gate[(size_t)(b * S_ + sq) * NH + h];
        float f = fv[v] * g;
        size_t rowb = (size_t)(b * S_ + sq) * DMODEL + h * DK;
#pragma unroll
        for (int t = 0; t < 4; ++t)
            Ob[rowb + t * 16 + col] = f2bf(o[t][v] * f);
    }
}

// ---------------------------------------------------------------------------
extern "C" void kernel_launch(void* const* d_in, const int* in_sizes, int n_in,
                              void* d_out, int out_size, void* d_ws, size_t ws_size,
                              hipStream_t stream) {
    (void)in_sizes; (void)n_in; (void)out_size; (void)ws_size;
    const float* x    = (const float*)d_in[0];  // [2,2048,1024]
    const float* w    = (const float*)d_in[1];  // [2560,1024]
    const float* qnw  = (const float*)d_in[2];  // [64]
    const float* knw  = (const float*)d_in[3];  // [64]
    const float* gw   = (const float*)d_in[4];  // [16,12]
    float* out = (float*)d_out;                 // [2,2048,1024]

    char* ws = (char*)d_ws;
    u16*   x_bf  = (u16*)ws;                 ws += (size_t)TOKENS * DMODEL * 2;      // 8 MB
    u16*   w_bf  = (u16*)ws;                 ws += (size_t)(QKV_N + DMODEL) * DMODEL * 2; // 5 MB
    float* qkv   = (float*)ws;               ws += (size_t)TOKENS * QKV_N * 4;       // 24 MB
    u16*   Qb    = (u16*)ws;                 ws += (size_t)B_ * NH  * S_ * DK * 2;   // 8 MB
    u16*   Kb    = (u16*)ws;                 ws += (size_t)B_ * NKV * S_ * DK * 2;   // 2 MB
    u16*   Vt    = (u16*)ws;                 ws += (size_t)B_ * NKV * DK * S_ * 2;   // 2 MB
    float* gate  = (float*)ws;               ws += (size_t)TOKENS * NH * 4;          // 256 KB
    u16*   a_bf  = (u16*)ws;                                                        // 8 MB

    // 1) conversions + gate
    {
        int n4 = TOKENS * DMODEL / 4;
        cvt_f32_bf16<<<n4 / 256, 256, 0, stream>>>(x, x_bf, n4);
        int w4 = (QKV_N + DMODEL) * DMODEL / 4;
        cvt_f32_bf16<<<w4 / 256, 256, 0, stream>>>(w, w_bf, w4);
        gate_kernel<<<TOKENS * NH / 256, 256, 0, stream>>>(x, gw, gate);
    }
    // 2) QKV projection: [4096,1536] = x_bf @ Wqkv^T
    gemm_bf16<<<dim3(QKV_N / 128, TOKENS / 64), 128, 0, stream>>>(
        x_bf, w_bf, qkv, TOKENS, QKV_N, DMODEL);
    // 3) RMSNorm + RoPE + V transpose
    norm_rope_kernel<<<TOKENS * 24 / 8, 256, 0, stream>>>(qkv, qnw, knw, Qb, Kb, Vt);
    // 4) causal GQA flash attention (+ gate), bf16 out
    attn_kernel<<<dim3(S_ / 64, B_ * NH), 128, 0, stream>>>(Qb, Kb, Vt, gate, a_bf);
    // 5) output projection: [4096,1024] = attn_bf @ Wo^T
    gemm_bf16<<<dim3(DMODEL / 128, TOKENS / 64), 128, 0, stream>>>(
        a_bf, w_bf + (size_t)QKV_N * DMODEL, out, TOKENS, DMODEL, DMODEL);
}
